// SigmaMoE_57054345560721
// MI455X (gfx1250) — matrix operand order repaired and checked by reading, not compiled, not run on hardware
//
#include <hip/hip_runtime.h>

#define Hdim 1024
#define Idim 512
#define NE   16
#define ISdim 1024
#define NT   4096            // B*S tokens
#define MT   64              // M tile (rows)
#define NTL  128             // N tile (cols)
#define KS   32              // K step (one WMMA)
#define LDK  40              // KS + 8 halves padding (even -> b32-aligned pairs)
#define TPE  64              // max 64-row tiles per expert (NT/MT)
#define MAXSLOTS 9216        // 8192 slots + 16*63 padding, rounded

typedef __attribute__((ext_vector_type(16))) _Float16 v16h;
typedef __attribute__((ext_vector_type(8)))  float    v8f;
typedef __attribute__((ext_vector_type(2)))  __fp16   v2fp;  // cvt_pkrtz result type
typedef __attribute__((ext_vector_type(4)))  __fp16   v4fp;
typedef int v4i __attribute__((vector_size(16)));
typedef _Float16 half_t;

// ---- async global->LDS copy (gfx1250), guarded so compile cannot break ----
#if defined(__has_builtin)
#if __has_builtin(__builtin_amdgcn_global_load_async_to_lds_b128) && \
    __has_builtin(__builtin_amdgcn_s_wait_asynccnt)
#define HAVE_ASYNC 1
#endif
#endif
#ifndef HAVE_ASYNC
#define HAVE_ASYNC 0
#endif

#if HAVE_ASYNC
typedef __attribute__((address_space(1))) v4i* gv4i_p;   // global int4*
typedef __attribute__((address_space(3))) v4i* lv4i_p;   // LDS int4*
#endif

// One 16-byte lane copy: per-lane global addr AND per-lane LDS addr, so the
// gathered-token copy maps directly onto GLOBAL_LOAD_ASYNC_TO_LDS_B128.
__device__ __forceinline__ void copy_b128_to_lds(half_t* lds, const half_t* g) {
#if HAVE_ASYNC
  __builtin_amdgcn_global_load_async_to_lds_b128(
      (gv4i_p)(void*)g, (lv4i_p)(void*)lds, 0, 0);
#else
  *reinterpret_cast<int4*>(lds) = *reinterpret_cast<const int4*>(g);
#endif
}

__device__ __forceinline__ void async_join() {
#if HAVE_ASYNC
  __builtin_amdgcn_s_wait_asynccnt(0);   // s_wait_asynccnt 0 before barrier
#endif
}

// ---- WMMA fragment loaders (16x16x32 f16, contiguous k order, wave32) ----

// A: 16x32 f16, row-major tile in LDS with stride LDK.
// lane<16: M=lane,    K = 0..15
// lane>=16: M=lane-16, K = 16..31
__device__ __forceinline__ v16h frag_a(const half_t* sm) {
  const int lane = threadIdx.x & 31;
  const int row  = lane & 15;
  const int kb   = (lane >> 4) << 4;       // 0 or 16
  const half_t* p = sm + row * LDK;
  v16h a;
#pragma unroll
  for (int j = 0; j < 16; ++j) a[j] = p[kb + j];
  return a;
}

// B: 32x16 f16 stored TRANSPOSED in LDS as [n][LDK] so each fragment
// is 16 contiguous halves per lane (2x ds_load_b128).
// lane<16: N=lane, K=0..15 ; lane>=16: N=lane-16, K=16..31
__device__ __forceinline__ v16h frag_bt(const half_t* smT) {
  const int lane = threadIdx.x & 31;
  const int col  = lane & 15;
  const int kb   = (lane >> 4) << 4;       // 0 or 16
  const half_t* p = smT + col * LDK + kb;
  v16h b;
#pragma unroll
  for (int j = 0; j < 16; ++j) b[j] = p[j];
  return b;
}

// Stage a 32xNTL f32 weight chunk into transposed-f16 LDS.
// Thread handles K-row pair (2*ri, 2*ri+1) x 8 cols: pack with cvt_pkrtz
// (one v_cvt_pk_rtz_f16_f32), one b32 LDS store per column.
__device__ __forceinline__ void stage_bt(half_t (*sBt)[LDK], const float* w,
                                         int ld, int ri, int c0) {
  const float* w0 = w + (size_t)(2 * ri) * ld + c0;
  const float* w1 = w0 + ld;
#pragma unroll
  for (int c = 0; c < 8; ++c) {
    v2fp pk = __builtin_amdgcn_cvt_pkrtz(w0[c], w1[c]);
    *reinterpret_cast<v2fp*>(&sBt[c0 + c][2 * ri]) = pk;
  }
}

__device__ __forceinline__ float fast_silu(float g) {
  return g * __builtin_amdgcn_rcpf(1.f + __expf(-g));
}

// ---------------- Router: softmax + top-2 ----------------
__global__ __launch_bounds__(256) void k_router(
    const float* __restrict__ x, const float* __restrict__ rw,
    int* __restrict__ tk_e, float* __restrict__ tk_w, int* __restrict__ count)
{
  const int lane = threadIdx.x & 31;
  const int t = blockIdx.x * 8 + (threadIdx.x >> 5);   // one wave per token
  const float* xr = x + (size_t)t * Hdim;
  float acc[NE];
#pragma unroll
  for (int e = 0; e < NE; ++e) acc[e] = 0.f;
  for (int h = lane; h < Hdim; h += 32) {
    const float xv = xr[h];
#pragma unroll
    for (int e = 0; e < NE; ++e) acc[e] += xv * rw[e * Hdim + h];
  }
#pragma unroll
  for (int e = 0; e < NE; ++e) {
#pragma unroll
    for (int off = 16; off > 0; off >>= 1) acc[e] += __shfl_xor(acc[e], off, 32);
  }
  if (lane == 0) {
    float mx = acc[0];
#pragma unroll
    for (int e = 1; e < NE; ++e) mx = fmaxf(mx, acc[e]);
    float p[NE];
#pragma unroll
    for (int e = 0; e < NE; ++e) p[e] = __expf(acc[e] - mx);
    int i0 = 0; float v0 = p[0];
#pragma unroll
    for (int e = 1; e < NE; ++e) if (p[e] > v0) { v0 = p[e]; i0 = e; }
    int i1 = (i0 == 0) ? 1 : 0; float v1 = p[i1];
#pragma unroll
    for (int e = 0; e < NE; ++e) if (e != i0 && p[e] > v1) { v1 = p[e]; i1 = e; }
    const float inv = 1.f / (v0 + v1);   // softmax denom cancels in top-2 renorm
    tk_e[t * 2]     = i0; tk_w[t * 2]     = v0 * inv;
    tk_e[t * 2 + 1] = i1; tk_w[t * 2 + 1] = v1 * inv;
    atomicAdd(&count[i0], 1);
    atomicAdd(&count[i1], 1);
  }
}

// Per-expert slot ranges, padded to MT so GEMM tiles never straddle experts.
__global__ void k_offsets(const int* __restrict__ count, int* __restrict__ po) {
  if (threadIdx.x == 0) {
    int acc = 0;
    for (int e = 0; e < NE; ++e) { po[e] = acc; acc += ((count[e] + MT - 1) / MT) * MT; }
    po[NE] = acc;
  }
}

__global__ __launch_bounds__(256) void k_scatter(
    const int* __restrict__ tk_e, const float* __restrict__ tk_w,
    const int* __restrict__ po, int* __restrict__ cursor,
    int* __restrict__ slot_token, float* __restrict__ slot_w)
{
  const int s = blockIdx.x * 256 + threadIdx.x;
  if (s >= NT * 2) return;
  const int e   = tk_e[s];
  const int pos = atomicAdd(&cursor[e], 1);
  const int idx = po[e] + pos;
  slot_token[idx] = s >> 1;
  slot_w[idx]     = tk_w[s];
}

__global__ __launch_bounds__(256) void k_cvt(const float* __restrict__ x,
                                             half_t* __restrict__ xh) {
  const int i = blockIdx.x * 256 + threadIdx.x;
  const float4 v = reinterpret_cast<const float4*>(x)[i];
  const v2fp lo = __builtin_amdgcn_cvt_pkrtz(v.x, v.y);
  const v2fp hi = __builtin_amdgcn_cvt_pkrtz(v.z, v.w);
  v4fp o; o[0] = lo[0]; o[1] = lo[1]; o[2] = hi[0]; o[3] = hi[1];
  reinterpret_cast<v4fp*>(xh)[i] = o;
}

// -------- Routed experts: gathered X @ {Wg,Wu}, fused silu*up*gate --------
// Wave blocking: wave owns 32 rows x 32 cols => 2 A-frags + 2 B-frags per
// matrix per K-step (vs 1A+4B before): LDS fragment reads cut by 1/3.
__global__ __launch_bounds__(256) void k_moe_gate_up(
    const half_t* __restrict__ xh,
    const float* __restrict__ Wg, const float* __restrict__ Wu,
    const int* __restrict__ count, const int* __restrict__ po,
    const int* __restrict__ slot_token, const float* __restrict__ slot_w,
    half_t* __restrict__ A)
{
  const int e  = blockIdx.x / TPE;
  const int mt = blockIdx.x % TPE;
  if (mt * MT >= count[e]) return;                    // block-uniform exit
  const int slotbase = po[e] + mt * MT;
  const int nbase    = blockIdx.y * NTL;
  const float* wg = Wg + (size_t)e * Hdim * Idim + nbase;
  const float* wu = Wu + (size_t)e * Hdim * Idim + nbase;

  __shared__ half_t sX [MT ][LDK];
  __shared__ half_t sGt[NTL][LDK];
  __shared__ half_t sUt[NTL][LDK];
  __shared__ int    sTok[MT];
  __shared__ float  sW [MT];

  const int tid = threadIdx.x;
  if (tid < MT) { sTok[tid] = slot_token[slotbase + tid]; sW[tid] = slot_w[slotbase + tid]; }
  __syncthreads();

  const int lane = tid & 31, wave = tid >> 5;
  const int mp = wave & 1;              // row half: rows mp*32 .. mp*32+31
  const int nq = wave >> 1;             // col quarter: cols nq*32 .. nq*32+31

  v8f accG[2][2], accU[2][2];
#pragma unroll
  for (int mi = 0; mi < 2; ++mi)
#pragma unroll
    for (int jj = 0; jj < 2; ++jj)
#pragma unroll
      for (int r = 0; r < 8; ++r) { accG[mi][jj][r] = 0.f; accU[mi][jj][r] = 0.f; }

  const int xr = tid >> 2, xc = (tid & 3) * 8;        // 64x32 X tile
  const int ri = tid & 15, c0 = (tid >> 4) * 8;       // K-pair x 8-col staging
  const int xtok = sTok[xr];

  for (int k0 = 0; k0 < Hdim; k0 += KS) {
    __syncthreads();
    copy_b128_to_lds(&sX[xr][xc], xh + (size_t)xtok * Hdim + k0 + xc);
    stage_bt(sGt, wg + (size_t)k0 * Idim, Idim, ri, c0);
    stage_bt(sUt, wu + (size_t)k0 * Idim, Idim, ri, c0);
    async_join();
    __syncthreads();
    v16h a[2];
#pragma unroll
    for (int mi = 0; mi < 2; ++mi) a[mi] = frag_a(&sX[(mp * 2 + mi) * 16][0]);
#pragma unroll
    for (int jj = 0; jj < 2; ++jj) {
      const v16h bg = frag_bt(&sGt[(nq * 2 + jj) * 16][0]);
      accG[0][jj] = __builtin_amdgcn_wmma_f32_16x16x32_f16(false, a[0], false, bg,
                                                           (short)0, accG[0][jj], false, false);
      accG[1][jj] = __builtin_amdgcn_wmma_f32_16x16x32_f16(false, a[1], false, bg,
                                                           (short)0, accG[1][jj], false, false);
      const v16h bu = frag_bt(&sUt[(nq * 2 + jj) * 16][0]);
      accU[0][jj] = __builtin_amdgcn_wmma_f32_16x16x32_f16(false, a[0], false, bu,
                                                           (short)0, accU[0][jj], false, false);
      accU[1][jj] = __builtin_amdgcn_wmma_f32_16x16x32_f16(false, a[1], false, bu,
                                                           (short)0, accU[1][jj], false, false);
    }
  }

  const int col = lane & 15, rb = (lane >> 4) * 8;
#pragma unroll
  for (int mi = 0; mi < 2; ++mi)
#pragma unroll
    for (int jj = 0; jj < 2; ++jj) {
      const int nj = nbase + (nq * 2 + jj) * 16 + col;
#pragma unroll
      for (int r = 0; r < 8; ++r) {
        const int m = (mp * 2 + mi) * 16 + rb + r;
        const float av = fast_silu(accG[mi][jj][r]) * accU[mi][jj][r] * sW[m];
        A[(size_t)(slotbase + m) * Idim + nj] = (half_t)av;
      }
    }
}

// -------- Routed down-projection: A @ Wd, atomic scatter-add to out --------
__global__ __launch_bounds__(256) void k_moe_down(
    const half_t* __restrict__ A, const float* __restrict__ Wd,
    const int* __restrict__ count, const int* __restrict__ po,
    const int* __restrict__ slot_token, float* __restrict__ out)
{
  const int e  = blockIdx.x / TPE;
  const int mt = blockIdx.x % TPE;
  if (mt * MT >= count[e]) return;
  const int slotbase = po[e] + mt * MT;
  const int nbase    = blockIdx.y * NTL;
  const float* wd = Wd + (size_t)e * Idim * Hdim + nbase;

  __shared__ half_t sA [MT ][LDK];
  __shared__ half_t sBt[NTL][LDK];
  __shared__ int    sTok[MT];

  const int tid = threadIdx.x;
  if (tid < MT) sTok[tid] = slot_token[slotbase + tid];
  __syncthreads();

  const int lane = tid & 31, wave = tid >> 5;
  const int mp = wave & 1, nq = wave >> 1;

  v8f acc[2][2];
#pragma unroll
  for (int mi = 0; mi < 2; ++mi)
#pragma unroll
    for (int jj = 0; jj < 2; ++jj)
#pragma unroll
      for (int r = 0; r < 8; ++r) acc[mi][jj][r] = 0.f;

  const int xr = tid >> 2, xc = (tid & 3) * 8;
  const int ri = tid & 15, c0 = (tid >> 4) * 8;

  for (int k0 = 0; k0 < Idim; k0 += KS) {
    __syncthreads();
    copy_b128_to_lds(&sA[xr][xc], A + (size_t)(slotbase + xr) * Idim + k0 + xc);
    stage_bt(sBt, wd + (size_t)k0 * Hdim, Hdim, ri, c0);
    async_join();
    __syncthreads();
    v16h a[2];
#pragma unroll
    for (int mi = 0; mi < 2; ++mi) a[mi] = frag_a(&sA[(mp * 2 + mi) * 16][0]);
#pragma unroll
    for (int jj = 0; jj < 2; ++jj) {
      const v16h b = frag_bt(&sBt[(nq * 2 + jj) * 16][0]);
      acc[0][jj] = __builtin_amdgcn_wmma_f32_16x16x32_f16(false, a[0], false, b,
                                                          (short)0, acc[0][jj], false, false);
      acc[1][jj] = __builtin_amdgcn_wmma_f32_16x16x32_f16(false, a[1], false, b,
                                                          (short)0, acc[1][jj], false, false);
    }
  }

  const int col = lane & 15, rb = (lane >> 4) * 8;
#pragma unroll
  for (int mi = 0; mi < 2; ++mi)
#pragma unroll
    for (int jj = 0; jj < 2; ++jj) {
      const int nj = nbase + (nq * 2 + jj) * 16 + col;
#pragma unroll
      for (int r = 0; r < 8; ++r) {
        const int m = (mp * 2 + mi) * 16 + rb + r;
        atomicAdd(&out[(size_t)sTok[m] * Hdim + nj], acc[mi][jj][r]);
      }
    }
}

// -------- Shared expert gate/up (dense, no gather) --------
__global__ __launch_bounds__(256) void k_shared_gate_up(
    const half_t* __restrict__ xh,
    const float* __restrict__ Wg, const float* __restrict__ Wu,
    half_t* __restrict__ A)
{
  const int rowbase = blockIdx.x * MT;
  const int nbase   = blockIdx.y * NTL;

  __shared__ half_t sX [MT ][LDK];
  __shared__ half_t sGt[NTL][LDK];
  __shared__ half_t sUt[NTL][LDK];

  const int tid = threadIdx.x;
  const int lane = tid & 31, wave = tid >> 5;
  const int mp = wave & 1, nq = wave >> 1;

  v8f accG[2][2], accU[2][2];
#pragma unroll
  for (int mi = 0; mi < 2; ++mi)
#pragma unroll
    for (int jj = 0; jj < 2; ++jj)
#pragma unroll
      for (int r = 0; r < 8; ++r) { accG[mi][jj][r] = 0.f; accU[mi][jj][r] = 0.f; }

  const int xr = tid >> 2, xc = (tid & 3) * 8;
  const int ri = tid & 15, c0 = (tid >> 4) * 8;

  for (int k0 = 0; k0 < Hdim; k0 += KS) {
    __syncthreads();
    copy_b128_to_lds(&sX[xr][xc], xh + (size_t)(rowbase + xr) * Hdim + k0 + xc);
    stage_bt(sGt, Wg + (size_t)k0 * ISdim + nbase, ISdim, ri, c0);
    stage_bt(sUt, Wu + (size_t)k0 * ISdim + nbase, ISdim, ri, c0);
    async_join();
    __syncthreads();
    v16h a[2];
#pragma unroll
    for (int mi = 0; mi < 2; ++mi) a[mi] = frag_a(&sX[(mp * 2 + mi) * 16][0]);
#pragma unroll
    for (int jj = 0; jj < 2; ++jj) {
      const v16h bg = frag_bt(&sGt[(nq * 2 + jj) * 16][0]);
      accG[0][jj] = __builtin_amdgcn_wmma_f32_16x16x32_f16(false, a[0], false, bg,
                                                           (short)0, accG[0][jj], false, false);
      accG[1][jj] = __builtin_amdgcn_wmma_f32_16x16x32_f16(false, a[1], false, bg,
                                                           (short)0, accG[1][jj], false, false);
      const v16h bu = frag_bt(&sUt[(nq * 2 + jj) * 16][0]);
      accU[0][jj] = __builtin_amdgcn_wmma_f32_16x16x32_f16(false, a[0], false, bu,
                                                           (short)0, accU[0][jj], false, false);
      accU[1][jj] = __builtin_amdgcn_wmma_f32_16x16x32_f16(false, a[1], false, bu,
                                                           (short)0, accU[1][jj], false, false);
    }
  }

  const int col = lane & 15, rb = (lane >> 4) * 8;
#pragma unroll
  for (int mi = 0; mi < 2; ++mi)
#pragma unroll
    for (int jj = 0; jj < 2; ++jj) {
      const int nj = nbase + (nq * 2 + jj) * 16 + col;
#pragma unroll
      for (int r = 0; r < 8; ++r) {
        const int m = (mp * 2 + mi) * 16 + rb + r;
        A[(size_t)(rowbase + m) * ISdim + nj] =
            (half_t)(fast_silu(accG[mi][jj][r]) * accU[mi][jj][r]);
      }
    }
}

// -------- Shared expert down: plain full write of out (runs first) --------
__global__ __launch_bounds__(256) void k_shared_down(
    const half_t* __restrict__ A, const float* __restrict__ Wd,
    float* __restrict__ out)
{
  const int rowbase = blockIdx.x * MT;
  const int nbase   = blockIdx.y * NTL;

  __shared__ half_t sA [MT ][LDK];
  __shared__ half_t sBt[NTL][LDK];

  const int tid = threadIdx.x;
  const int lane = tid & 31, wave = tid >> 5;
  const int mp = wave & 1, nq = wave >> 1;

  v8f acc[2][2];
#pragma unroll
  for (int mi = 0; mi < 2; ++mi)
#pragma unroll
    for (int jj = 0; jj < 2; ++jj)
#pragma unroll
      for (int r = 0; r < 8; ++r) acc[mi][jj][r] = 0.f;

  const int xr = tid >> 2, xc = (tid & 3) * 8;
  const int ri = tid & 15, c0 = (tid >> 4) * 8;

  for (int k0 = 0; k0 < ISdim; k0 += KS) {
    __syncthreads();
    copy_b128_to_lds(&sA[xr][xc], A + (size_t)(rowbase + xr) * ISdim + k0 + xc);
    stage_bt(sBt, Wd + (size_t)k0 * Hdim + nbase, Hdim, ri, c0);
    async_join();
    __syncthreads();
    v16h a[2];
#pragma unroll
    for (int mi = 0; mi < 2; ++mi) a[mi] = frag_a(&sA[(mp * 2 + mi) * 16][0]);
#pragma unroll
    for (int jj = 0; jj < 2; ++jj) {
      const v16h b = frag_bt(&sBt[(nq * 2 + jj) * 16][0]);
      acc[0][jj] = __builtin_amdgcn_wmma_f32_16x16x32_f16(false, a[0], false, b,
                                                          (short)0, acc[0][jj], false, false);
      acc[1][jj] = __builtin_amdgcn_wmma_f32_16x16x32_f16(false, a[1], false, b,
                                                          (short)0, acc[1][jj], false, false);
    }
  }

  const int col = lane & 15, rb = (lane >> 4) * 8;
#pragma unroll
  for (int mi = 0; mi < 2; ++mi)
#pragma unroll
    for (int jj = 0; jj < 2; ++jj) {
      const int nj = nbase + (nq * 2 + jj) * 16 + col;
#pragma unroll
      for (int r = 0; r < 8; ++r) {
        const int m = (mp * 2 + mi) * 16 + rb + r;
        out[(size_t)(rowbase + m) * Hdim + nj] = acc[mi][jj][r];
      }
    }
}

extern "C" void kernel_launch(void* const* d_in, const int* in_sizes, int n_in,
                              void* d_out, int out_size, void* d_ws, size_t ws_size,
                              hipStream_t stream) {
  (void)in_sizes; (void)n_in; (void)out_size; (void)ws_size;
  const float* x  = (const float*)d_in[0];
  const float* rw = (const float*)d_in[1];
  const float* eg = (const float*)d_in[2];
  const float* eu = (const float*)d_in[3];
  const float* ed = (const float*)d_in[4];
  const float* sg = (const float*)d_in[5];
  const float* su = (const float*)d_in[6];
  const float* sd = (const float*)d_in[7];
  float* out = (float*)d_out;

  char* ws = (char*)d_ws;
  size_t off = 0;
  auto carve = [&](size_t bytes) -> void* {
    void* p = ws + off;
    off += (bytes + 255) & ~(size_t)255;
    return p;
  };
  half_t* xh    = (half_t*)carve((size_t)NT * Hdim * 2);       // f16 activations
  half_t* Ar    = (half_t*)carve((size_t)MAXSLOTS * Idim * 2); // routed intermediate
  half_t* As    = (half_t*)carve((size_t)NT * ISdim * 2);      // shared intermediate
  int*   tk_e   = (int*)  carve(NT * 2 * sizeof(int));
  float* tk_w   = (float*)carve(NT * 2 * sizeof(float));
  int*   count  = (int*)  carve(NE * sizeof(int));
  int*   po     = (int*)  carve((NE + 1) * sizeof(int));
  int*   cursor = (int*)  carve(NE * sizeof(int));
  int*   stok   = (int*)  carve(MAXSLOTS * sizeof(int));
  float* sw     = (float*)carve(MAXSLOTS * sizeof(float));

  // zero counters + slot arrays (padded slots -> token 0, weight 0 -> inert)
  (void)hipMemsetAsync(count,  0, NE * sizeof(int), stream);
  (void)hipMemsetAsync(cursor, 0, NE * sizeof(int), stream);
  (void)hipMemsetAsync(stok,   0, MAXSLOTS * sizeof(int), stream);
  (void)hipMemsetAsync(sw,     0, MAXSLOTS * sizeof(float), stream);

  k_cvt    <<<(NT * Hdim) / (256 * 4), 256, 0, stream>>>(x, xh);
  k_router <<<NT / 8, 256, 0, stream>>>(x, rw, tk_e, tk_w, count);
  k_offsets<<<1, 32, 0, stream>>>(count, po);
  k_scatter<<<(NT * 2) / 256, 256, 0, stream>>>(tk_e, tk_w, po, cursor, stok, sw);

  // shared expert (writes every element of out) ...
  k_shared_gate_up<<<dim3(NT / MT, ISdim / NTL), 256, 0, stream>>>(xh, sg, su, As);
  k_shared_down   <<<dim3(NT / MT, Hdim  / NTL), 256, 0, stream>>>(As, sd, out);

  // ... then routed experts accumulate on top via atomics
  k_moe_gate_up<<<dim3(NE * TPE, Idim / NTL), 256, 0, stream>>>(
      xh, eg, eu, count, po, stok, sw, Ar);
  k_moe_down   <<<dim3(NE * TPE, Hdim / NTL), 256, 0, stream>>>(
      Ar, ed, count, po, stok, out);
}
